// NeoNet_13134009991199
// MI455X (gfx1250) — matrix operand order repaired and checked by reading, not compiled, not run on hardware
//
#include <hip/hip_runtime.h>
#include <hip/hip_bf16.h>
#include <math.h>

#define BATCH 64
#define NNODE 1025
#define MNODE 1024
#define TFEAT 96
#define QVECN 95
#define INCH  128
#define EPSV  1e-5f

typedef __attribute__((ext_vector_type(16))) __bf16 bf16x16;
typedef __attribute__((ext_vector_type(8)))  __bf16 bf16x8;
typedef __attribute__((ext_vector_type(8)))  float  floatx8;

__device__ __forceinline__ __bf16 f2bf(float x) {
  unsigned u = __builtin_bit_cast(unsigned, x);
  unsigned r = u + 0x7fffu + ((u >> 16) & 1u);
  unsigned short h = (unsigned short)(r >> 16);
  return __builtin_bit_cast(__bf16, h);
}

// CDNA5 async global->LDS copy (ASYNCcnt-tracked, bypasses VGPRs).
__device__ __forceinline__ void async_lds_b128(unsigned lds_off,
                                               unsigned long long gaddr) {
  asm volatile("global_load_async_to_lds_b128 %0, %1, off"
               :: "v"(lds_off), "v"(gaddr) : "memory");
}
__device__ __forceinline__ void wait_asynccnt0() {
  asm volatile("s_wait_asynccnt 0x0" ::: "memory");
}

// ---------------------------------------------------------------------------
// q_conv: 4 chained linears on (64 x 95), all in one block via LDS ping-pong.
// ---------------------------------------------------------------------------
__global__ __launch_bounds__(256) void qconv_kernel(
    const float* __restrict__ qv,
    const float* __restrict__ W1, const float* __restrict__ b1,
    const float* __restrict__ W2, const float* __restrict__ b2,
    const float* __restrict__ W3, const float* __restrict__ b3,
    const float* __restrict__ W4, const float* __restrict__ b4,
    float* __restrict__ qout)
{
  __shared__ float bufA[64 * 128];
  __shared__ float bufB[64 * 128];
  const int tid = threadIdx.x;
  for (int i = tid; i < 64 * QVECN; i += 256) bufA[i] = qv[i];
  __syncthreads();
  for (int i = tid; i < 64 * 64; i += 256) {           // L1: 95 -> 64
    int r = i >> 6, c = i & 63;
    float s = b1[c];
    for (int k = 0; k < QVECN; ++k) s += bufA[r * QVECN + k] * W1[k * 64 + c];
    bufB[r * 64 + c] = s;
  }
  __syncthreads();
  for (int i = tid; i < 64 * 128; i += 256) {          // L2: 64 -> 128
    int r = i >> 7, c = i & 127;
    float s = b2[c];
    for (int k = 0; k < 64; ++k) s += bufB[r * 64 + k] * W2[k * 128 + c];
    bufA[r * 128 + c] = s;
  }
  __syncthreads();
  for (int i = tid; i < 64 * 64; i += 256) {           // L3: 128 -> 64
    int r = i >> 6, c = i & 63;
    float s = b3[c];
    for (int k = 0; k < 128; ++k) s += bufA[r * 128 + k] * W3[k * 64 + c];
    bufB[r * 64 + c] = s;
  }
  __syncthreads();
  for (int i = tid; i < 64 * 32; i += 256) {           // L4: 64 -> 32
    int r = i >> 5, c = i & 31;
    float s = b4[c];
    for (int k = 0; k < 64; ++k) s += bufB[r * 64 + k] * W4[k * 32 + c];
    qout[i] = s;
  }
}

// ---------------------------------------------------------------------------
// Repack conv weight (Cout, Cin, 3) f32 -> B matrix (3*Cin x Cout) bf16,
// with K index k = child*Cin + f  so one K-chunk of 32 stays inside a child.
// ---------------------------------------------------------------------------
__global__ __launch_bounds__(256) void pack_w_kernel(
    const float* __restrict__ Wc, __bf16* __restrict__ Wb, int Cin, int Cout)
{
  int i = blockIdx.x * 256 + threadIdx.x;
  int total = 3 * Cin * Cout;
  if (i >= total) return;
  int k = i / Cout, o = i - k * Cout;
  int child = k / Cin, f = k - child * Cin;
  Wb[i] = f2bf(Wc[(o * Cin + f) * 3 + child]);
}

// ---------------------------------------------------------------------------
// Build augmented activation A0 (B x N x 128) bf16 = [tree feats | q bcast]
// ---------------------------------------------------------------------------
__global__ __launch_bounds__(256) void pack_aug_kernel(
    const float* __restrict__ trees, const float* __restrict__ qout,
    __bf16* __restrict__ A0)
{
  size_t i = (size_t)blockIdx.x * 256 + threadIdx.x;
  const size_t total = (size_t)BATCH * NNODE * INCH;
  if (i >= total) return;
  int c = (int)(i & (INCH - 1));
  size_t t = i >> 7;
  int n = (int)(t % NNODE);
  int b = (int)(t / NNODE);
  float v = (c < TFEAT) ? trees[((size_t)b * NNODE + n) * TFEAT + c]
                        : qout[b * 32 + (c - TFEAT)];
  A0[i] = f2bf(v);
}

// ---------------------------------------------------------------------------
// Fused gather + bf16 WMMA GEMM with double-buffered async global->LDS.
//   Y[(b,m), o] = sum_{child,f} act[b, idx[b,3m+child], f] * Wb[child*Cin+f, o]
// Block: 256 threads (8 waves). Tile: 128 rows x 64 cols, K-step 32.
// ---------------------------------------------------------------------------
__global__ __launch_bounds__(256) void tree_conv_gemm(
    const __bf16* __restrict__ act,   // B x N x Cin (bf16)
    const __bf16* __restrict__ Wb,    // (3*Cin) x Cout (bf16)
    const float* __restrict__ bias,   // Cout
    const int*   __restrict__ idx,    // B x 3M
    float*       __restrict__ Y,      // (B*M) x Cout (f32)
    int Cin, int lc, int Cout)
{
  // padded LDS strides to spread banks (A: 40 elems/row, B: 72 elems/row)
  __shared__ __align__(32) __bf16 As[2][128 * 40];
  __shared__ __align__(32) __bf16 Bs[2][32 * 72];
  const unsigned A_STRIDE = 128 * 40 * 2;  // bytes per A buffer
  const unsigned B_STRIDE = 32 * 72 * 2;   // bytes per B buffer

  const int tid  = threadIdx.x;
  const int wave = tid >> 5;
  const int lane = tid & 31;
  const int R0 = blockIdx.x * 128;           // output row base (same b for block)
  const int C0 = blockIdx.y * 64;            // output col base
  const int b  = R0 >> 10;                   // M = 1024
  const int m0 = R0 & (MNODE - 1);

  const int rA = tid >> 1;                   // A staging: 2 threads per row
  const int hA = tid & 1;
  const int kB = tid >> 3;                   // B staging: 8 threads per K-row
  const int sB = tid & 7;
  const int* idxRow = idx + (size_t)b * 3 * MNODE + 3 * (m0 + rA);
  const __bf16* actB = act + (size_t)b * NNODE * Cin;
  const __bf16* wSrc = Wb + (size_t)kB * Cout + C0 + sB * 8;

  // LDS byte offsets for the async DMA destinations (low 32 bits of the
  // generic address are the wave-relative LDS offset on CDNA5).
  const unsigned aDst0 = (unsigned)(uintptr_t)&As[0][rA * 40] + (unsigned)hA * 32u;
  const unsigned bDst0 = (unsigned)(uintptr_t)&Bs[0][kB * 72 + sB * 8];

  floatx8 acc0 = {}, acc1 = {}, acc2 = {}, acc3 = {};

  // WMMA A fragment addressing (16x32 bf16, ISA layout):
  //  lanes 0-15: row = lane,    K = {0..7, 16..23}
  //  lanes 16-31: row = lane-16, K = {8..15, 24..31}
  const int arow = wave * 16 + (lane & 15);
  const int kofs = (lane & 16) ? 8 : 0;

  const int K = 3 * Cin;

  auto stage = [&](int k0, int bufSel) {
    const int child = k0 >> lc;              // which of the 3 children
    const int kin   = k0 & (Cin - 1);        // feature offset within child
    const int node  = idxRow[child];
    const unsigned long long ga =
        (unsigned long long)(uintptr_t)(actB + (size_t)node * Cin + kin) +
        (unsigned long long)hA * 32ull;
    const unsigned long long gw =
        (unsigned long long)(uintptr_t)(wSrc + (size_t)k0 * Cout);
    const unsigned ad = aDst0 + (unsigned)bufSel * A_STRIDE;
    const unsigned bd = bDst0 + (unsigned)bufSel * B_STRIDE;
    async_lds_b128(ad, ga);
    async_lds_b128(ad + 16u, ga + 16ull);
    async_lds_b128(bd, gw);
  };

  stage(0, 0);
  int buf = 0;
  for (int k0 = 0; k0 < K; k0 += 32) {
    wait_asynccnt0();    // our async loads for As[buf]/Bs[buf] have landed
    __syncthreads();     // everyone's loads landed; previous compute finished
    if (k0 + 32 < K) stage(k0 + 32, buf ^ 1);   // prefetch next tile

    const __bf16* Ab = &As[buf][0];
    const __bf16* Bb = &Bs[buf][0];

    bf16x8 alo = *reinterpret_cast<const bf16x8*>(&Ab[arow * 40 + kofs]);
    bf16x8 ahi = *reinterpret_cast<const bf16x8*>(&Ab[arow * 40 + kofs + 16]);
    bf16x16 afrag = __builtin_shufflevector(alo, ahi,
        0, 1, 2, 3, 4, 5, 6, 7, 8, 9, 10, 11, 12, 13, 14, 15);

    // Load all four B fragments first so the WMMAs can issue back-to-back.
    bf16x16 bf0, bf1, bf2, bf3;
#pragma unroll
    for (int s = 0; s < 16; ++s) {
      const int kk = kofs + (s & 7) + ((s & 8) ? 16 : 0);
      const __bf16* bp = &Bb[kk * 72 + (lane & 15)];
      bf0[s] = bp[0];
      bf1[s] = bp[16];
      bf2[s] = bp[32];
      bf3[s] = bp[48];
    }

    acc0 = __builtin_amdgcn_wmma_f32_16x16x32_bf16(false, afrag, false, bf0,
                                                   (short)0, acc0, false, false);
    acc1 = __builtin_amdgcn_wmma_f32_16x16x32_bf16(false, afrag, false, bf1,
                                                   (short)0, acc1, false, false);
    acc2 = __builtin_amdgcn_wmma_f32_16x16x32_bf16(false, afrag, false, bf2,
                                                   (short)0, acc2, false, false);
    acc3 = __builtin_amdgcn_wmma_f32_16x16x32_bf16(false, afrag, false, bf3,
                                                   (short)0, acc3, false, false);
    buf ^= 1;
  }

  // C/D layout: VGPR v, lanes 0-15 -> M=v, N=lane; lanes 16-31 -> M=v+8.
  const int row0 = R0 + wave * 16 + ((lane & 16) ? 8 : 0);
  const int col  = C0 + (lane & 15);
  const float bb0 = bias[col],      bb1 = bias[col + 16];
  const float bb2 = bias[col + 32], bb3 = bias[col + 48];
#pragma unroll
  for (int v = 0; v < 8; ++v) {
    float* yr = Y + (size_t)(row0 + v) * Cout + col;
    yr[0]  = acc0[v] + bb0;
    yr[16] = acc1[v] + bb1;
    yr[32] = acc2[v] + bb2;
    yr[48] = acc3[v] + bb3;
  }
}

// ---------------------------------------------------------------------------
// Per-sample sum / sumsq over Y (B*M x C), two-stage deterministic reduce.
// ---------------------------------------------------------------------------
__global__ __launch_bounds__(256) void stats_partial(
    const float* __restrict__ Y, float* __restrict__ part, int C, int chunks)
{
  const int b = blockIdx.y;
  const int ch = blockIdx.x;
  const int rowsPer = MNODE / chunks;
  const float* p = Y + ((size_t)b * MNODE + (size_t)ch * rowsPer) * C;
  const size_t per = (size_t)rowsPer * C;
  float s = 0.f, s2 = 0.f;
  for (size_t i = threadIdx.x; i < per; i += 256) {
    float v = p[i];
    s += v;
    s2 += v * v;
  }
  __shared__ float rs[256], rq[256];
  rs[threadIdx.x] = s;
  rq[threadIdx.x] = s2;
  __syncthreads();
  for (int o = 128; o > 0; o >>= 1) {
    if (threadIdx.x < o) {
      rs[threadIdx.x] += rs[threadIdx.x + o];
      rq[threadIdx.x] += rq[threadIdx.x + o];
    }
    __syncthreads();
  }
  if (threadIdx.x == 0) {
    part[(b * chunks + ch) * 2 + 0] = rs[0];
    part[(b * chunks + ch) * 2 + 1] = rq[0];
  }
}

__global__ void stats_final(const float* __restrict__ part,
                            float* __restrict__ stats, int C, int chunks)
{
  int b = threadIdx.x;
  if (b >= BATCH) return;
  float s = 0.f, s2 = 0.f;
  for (int i = 0; i < chunks; ++i) {
    s  += part[(b * chunks + i) * 2 + 0];
    s2 += part[(b * chunks + i) * 2 + 1];
  }
  // zero node contributes 0 to sums but C elements to the count
  float count = (float)C * (float)NNODE;
  float mean = s / count;
  float var  = (s2 - s * s / count) / (count - 1.0f);   // ddof=1
  float inv  = 1.0f / (sqrtf(fmaxf(var, 0.f)) + EPSV);
  stats[2 * b + 0] = mean;
  stats[2 * b + 1] = inv;
}

// ---------------------------------------------------------------------------
// norm + leaky-ReLU + bf16 cast -> next activation (B x N x C), node0 = zeros
// ---------------------------------------------------------------------------
__global__ __launch_bounds__(256) void normalize_act(
    const float* __restrict__ Y, const float* __restrict__ stats,
    __bf16* __restrict__ A, int C, int lcc, float slope)
{
  size_t i = (size_t)blockIdx.x * 256 + threadIdx.x;
  const size_t total = ((size_t)BATCH * NNODE) << lcc;
  if (i >= total) return;
  int c = (int)(i & (C - 1));
  size_t t = i >> lcc;
  int n = (int)(t % NNODE);
  int b = (int)(t / NNODE);
  float mean = stats[2 * b], inv = stats[2 * b + 1];
  float v = (n == 0) ? 0.f : Y[((size_t)b * MNODE + (n - 1)) * C + c];
  v = (v - mean) * inv;
  v = (v >= 0.f) ? v : slope * v;
  A[i] = f2bf(v);
}

// ---------------------------------------------------------------------------
// DynamicPooling: norm is monotone -> max pre-norm (vs node0 = 0), norm once.
// ---------------------------------------------------------------------------
__global__ __launch_bounds__(128) void final_max(
    const float* __restrict__ Y3, const float* __restrict__ stats,
    float* __restrict__ out)
{
  const int b = blockIdx.x, c = threadIdx.x;
  float mx = 0.f;  // node 0 pre-norm value
  const float* p = Y3 + (size_t)b * MNODE * 128 + c;
  for (int m = 0; m < MNODE; ++m) mx = fmaxf(mx, p[(size_t)m * 128]);
  out[b * 128 + c] = (mx - stats[2 * b]) * stats[2 * b + 1];
}

// ---------------------------------------------------------------------------
extern "C" void kernel_launch(void* const* d_in, const int* in_sizes, int n_in,
                              void* d_out, int out_size, void* d_ws, size_t ws_size,
                              hipStream_t stream) {
  (void)in_sizes; (void)n_in; (void)out_size; (void)ws_size;
  const float* q_vecs = (const float*)d_in[0];
  const float* trees  = (const float*)d_in[1];
  const int*   idxes  = (const int*)d_in[2];
  const float* Wq1 = (const float*)d_in[3],  *bq1 = (const float*)d_in[4];
  const float* Wq2 = (const float*)d_in[5],  *bq2 = (const float*)d_in[6];
  const float* Wq3 = (const float*)d_in[7],  *bq3 = (const float*)d_in[8];
  const float* Wq4 = (const float*)d_in[9],  *bq4 = (const float*)d_in[10];
  const float* Wc1 = (const float*)d_in[11], *bc1 = (const float*)d_in[12];
  const float* Wc2 = (const float*)d_in[13], *bc2 = (const float*)d_in[14];
  const float* Wc3 = (const float*)d_in[15], *bc3 = (const float*)d_in[16];

  char* ws = (char*)d_ws;
  size_t off = 0;
  auto alloc = [&](size_t bytes) -> void* {
    void* p = ws + off;
    off += (bytes + 255) & ~(size_t)255;
    return p;
  };
  float*  qout  = (float*) alloc((size_t)64 * 32 * 4);
  __bf16* Wb1   = (__bf16*)alloc((size_t)384 * 512 * 2);
  __bf16* Wb2   = (__bf16*)alloc((size_t)1536 * 256 * 2);
  __bf16* Wb3   = (__bf16*)alloc((size_t)768 * 128 * 2);
  float*  stats = (float*) alloc((size_t)BATCH * 2 * 4);
  float*  part  = (float*) alloc((size_t)BATCH * 16 * 2 * 4);
  // region A: A0 (16.8MB) then A2 (33.6MB)  [A0 dead before A2 written]
  __bf16* regA  = (__bf16*)alloc((size_t)BATCH * NNODE * 256 * 2);
  __bf16* A0 = regA;
  __bf16* A2 = regA;
  // region B: A1 (67.2MB)
  __bf16* A1 = (__bf16*)alloc((size_t)BATCH * NNODE * 512 * 2);
  // region C: Y1 (134MB) -> Y2 (67MB @0, after Y1 dead) ; Y3 (33.5MB @ +67MB)
  float* regC = (float*)alloc((size_t)BATCH * MNODE * 512 * 4);
  float* Y1 = regC;
  float* Y2 = regC;
  float* Y3 = (float*)((char*)regC + (size_t)BATCH * MNODE * 256 * 4);

  const int GR = (BATCH * MNODE) / 128;     // 512 row-tiles

  qconv_kernel<<<1, 256, 0, stream>>>(q_vecs, Wq1, bq1, Wq2, bq2, Wq3, bq3,
                                      Wq4, bq4, qout);
  pack_w_kernel<<<(3 * 128 * 512 + 255) / 256, 256, 0, stream>>>(Wc1, Wb1, 128, 512);
  pack_w_kernel<<<(3 * 512 * 256 + 255) / 256, 256, 0, stream>>>(Wc2, Wb2, 512, 256);
  pack_w_kernel<<<(3 * 256 * 128 + 255) / 256, 256, 0, stream>>>(Wc3, Wb3, 256, 128);
  {
    size_t tot = (size_t)BATCH * NNODE * INCH;
    pack_aug_kernel<<<(unsigned)((tot + 255) / 256), 256, 0, stream>>>(trees, qout, A0);
  }
  // Layer 1: Cin=128 (lc=7), Cout=512
  tree_conv_gemm<<<dim3(GR, 8), 256, 0, stream>>>(A0, Wb1, bc1, idxes, Y1, 128, 7, 512);
  stats_partial<<<dim3(16, BATCH), 256, 0, stream>>>(Y1, part, 512, 16);
  stats_final<<<1, 64, 0, stream>>>(part, stats, 512, 16);
  {
    size_t tot = (size_t)BATCH * NNODE * 512;
    normalize_act<<<(unsigned)((tot + 255) / 256), 256, 0, stream>>>(Y1, stats, A1, 512, 9, 0.01f);
  }
  // Layer 2: Cin=512 (lc=9), Cout=256
  tree_conv_gemm<<<dim3(GR, 4), 256, 0, stream>>>(A1, Wb2, bc2, idxes, Y2, 512, 9, 256);
  stats_partial<<<dim3(16, BATCH), 256, 0, stream>>>(Y2, part, 256, 16);
  stats_final<<<1, 64, 0, stream>>>(part, stats, 256, 16);
  {
    size_t tot = (size_t)BATCH * NNODE * 256;
    normalize_act<<<(unsigned)((tot + 255) / 256), 256, 0, stream>>>(Y2, stats, A2, 256, 8, 0.01f);
  }
  // Layer 3: Cin=256 (lc=8), Cout=128
  tree_conv_gemm<<<dim3(GR, 2), 256, 0, stream>>>(A2, Wb3, bc3, idxes, Y3, 256, 8, 128);
  stats_partial<<<dim3(16, BATCH), 256, 0, stream>>>(Y3, part, 128, 16);
  stats_final<<<1, 64, 0, stream>>>(part, stats, 128, 16);
  final_max<<<BATCH, 128, 0, stream>>>(Y3, stats, (float*)d_out);
}
